// _MultiHeadAttention_704374637105
// MI455X (gfx1250) — matrix operand order repaired and checked by reading, not compiled, not run on hardware
//
#include <hip/hip_runtime.h>
#include <hip/hip_bf16.h>

// Problem constants (match reference)
#define D_MODEL   1024
#define NUM_HEADS 16
#define DEPTH     64
#define BATCH     2
#define SEQ       2048

typedef __bf16 bf16_t;
typedef __attribute__((ext_vector_type(16))) __bf16 v16bf;
typedef __attribute__((ext_vector_type(8)))  float  v8f;
typedef int v2i_t __attribute__((vector_size(2 * sizeof(int))));

// GEMM tiling: 128 threads = 4 waves (wave32). Block tile 64x64, K-step 32.
#define BM 64
#define BN 64
#define BK 32
#define LDT 36        // GEMM LDS row stride in bf16 (18*M mod 64 all distinct -> no conflicts)
#define NTHREADS 128

// Fused attention kernel LDS geometry
#define LSTR  2052    // logits row stride in floats (2052 mod 64 = 4 -> conflict-free b64 reads)
#define QKSTR 68      // Q/K/V LDS row stride in bf16 (34*M mod 64 all distinct)
#define SMEM_FUSED (16 * LSTR * 4 + 16 * QKSTR * 2 + 64 * QKSTR * 2 + 128 * 4 + 16 * 4)

// ---------------------------------------------------------------------------
// Async global->LDS copy (CDNA5 GLOBAL_LOAD_ASYNC_TO_LDS_B64, ASYNCcnt).
// Builtin signature (from compiler diagnostic): param0 = v2i addrspace(1)*,
// param1 = v2i addrspace(3)*, then (imm offset, imm cpol).
// ---------------------------------------------------------------------------
#if defined(__gfx1250__) && __has_builtin(__builtin_amdgcn_global_load_async_to_lds_b64)
#define HAVE_ASYNC_LDS 1
#else
#define HAVE_ASYNC_LDS 0
#endif

__device__ __forceinline__ void cp_b64_to_lds(void* ldsdst, const void* gsrc) {
#if HAVE_ASYNC_LDS
  __builtin_amdgcn_global_load_async_to_lds_b64(
      (__attribute__((address_space(1))) v2i_t*)gsrc,
      (__attribute__((address_space(3))) v2i_t*)ldsdst, 0, 0);
#else
  *(unsigned long long*)ldsdst = *(const unsigned long long*)gsrc;
#endif
}

__device__ __forceinline__ void cp_to_lds_wait() {
#if HAVE_ASYNC_LDS
  asm volatile("s_wait_asynccnt 0x0" ::: "memory");
#endif
}

// ---------------------------------------------------------------------------
// Fragment load from LDS (row-major, K contiguous). Per CDNA5 ISA 7.12.2:
// 16-bit A 16x32: lane L holds M=L%16, half=L/16; VGPR r holds K-pair
//   kbase(r,half) = (r<4 ? 2r : 2r+8) + 8*half  ->  dword index = (r<4?r:r+4)+4*half
// B 32x16 is symmetric with N in place of M when staged as [N][K].
// ---------------------------------------------------------------------------
__device__ __forceinline__ v16bf load_frag(const bf16_t* row, int half) {
  union { unsigned u[8]; v16bf v; } f;
  const unsigned* p = (const unsigned*)row;
  const int h4 = half * 4;
#pragma unroll
  for (int r = 0; r < 8; ++r) {
    const int d = (r < 4 ? r : r + 4) + h4;
    f.u[r] = p[d];
  }
  return f.v;
}

// Build an A-fragment from an fp32 row in LDS (reads b64 pairs, packs to bf16).
__device__ __forceinline__ v16bf frag_from_f32(const float* rowbase, int half) {
  union { unsigned u[8]; v16bf v; } f;
  const int h8 = half * 8;
#pragma unroll
  for (int r = 0; r < 8; ++r) {
    const int kw = (r < 4 ? 2 * r : 2 * r + 8) + h8;
    union { bf16_t h[2]; unsigned uu; } pk;
    pk.h[0] = (bf16_t)rowbase[kw];
    pk.h[1] = (bf16_t)rowbase[kw + 1];
    f.u[r] = pk.uu;
  }
  return f.v;
}

// Stage a 64x32 fp32 tile (row-major source, leading dim ld) into LDS as bf16 [row][k].
__device__ __forceinline__ void stage_rowmajor(bf16_t* lds, const float* src, int ld,
                                               int r0, int k0) {
  const int t = threadIdx.x;
#pragma unroll
  for (int e = 0; e < 16; ++e) {
    const int idx = t * 16 + e;       // 0..2047
    const int r = idx >> 5;
    const int k = idx & 31;
    lds[r * LDT + k] = (bf16_t)src[(size_t)(r0 + r) * ld + (k0 + k)];
  }
}

// Stage a 32x64 fp32 tile from a [K][N] row-major source into LDS as [n][k] (transpose).
__device__ __forceinline__ void stage_transpose(bf16_t* lds, const float* src, int ld,
                                                int k0, int n0) {
  const int t = threadIdx.x;
#pragma unroll
  for (int e = 0; e < 16; ++e) {
    const int idx = t * 16 + e;
    const int k = idx >> 6;
    const int n = idx & 63;
    lds[n * LDT + k] = (bf16_t)src[(size_t)(k0 + k) * ld + (n0 + n)];
  }
}

// One BK=32 step of the 4-wave 64x64 block tile: 4 WMMAs per wave.
__device__ __forceinline__ void wave_mma(const bf16_t* ldsA, const bf16_t* ldsB, v8f acc[4]) {
  const int lane = threadIdx.x & 31;
  const int wave = threadIdx.x >> 5;
  const int wm = (wave >> 1) * 32;
  const int wn = (wave & 1) * 32;
  const int rl = lane & 15;
  const int half = lane >> 4;
  v16bf a0 = load_frag(ldsA + (wm + rl) * LDT, half);
  v16bf a1 = load_frag(ldsA + (wm + 16 + rl) * LDT, half);
  v16bf b0 = load_frag(ldsB + (wn + rl) * LDT, half);
  v16bf b1 = load_frag(ldsB + (wn + 16 + rl) * LDT, half);
  acc[0] = __builtin_amdgcn_wmma_f32_16x16x32_bf16(false, a0, false, b0, (short)0, acc[0], false, false);
  acc[1] = __builtin_amdgcn_wmma_f32_16x16x32_bf16(false, a0, false, b1, (short)0, acc[1], false, false);
  acc[2] = __builtin_amdgcn_wmma_f32_16x16x32_bf16(false, a1, false, b0, (short)0, acc[2], false, false);
  acc[3] = __builtin_amdgcn_wmma_f32_16x16x32_bf16(false, a1, false, b1, (short)0, acc[3], false, false);
}

#define ACC_ZERO { {0,0,0,0,0,0,0,0}, {0,0,0,0,0,0,0,0}, {0,0,0,0,0,0,0,0}, {0,0,0,0,0,0,0,0} }

// C/D layout (7.12.2): element rr of v8f at lane L -> M = rr + 8*(L/16), N = L%16.
#define EPILOGUE_VARS                                 \
  const int lane = threadIdx.x & 31;                  \
  const int wave = threadIdx.x >> 5;                  \
  const int wm = (wave >> 1) * 32;                    \
  const int wn = (wave & 1) * 32;                     \
  const int nl = lane & 15;                           \
  const int halfc = lane >> 4;

// ---------------------------------------------------------------------------
// Kernel 1: QKV projections -> bf16. grid = (D_MODEL/BN, B*S/BM, 3)
//   z=0: Qh[b][h][s][d]   z=1: Kh[b][h][s][d]   z=2: VhT[b][h][d][s]
// ---------------------------------------------------------------------------
__global__ __launch_bounds__(NTHREADS)
void proj_qkv_kernel(const float* __restrict__ q, const float* __restrict__ k,
                     const float* __restrict__ v,
                     const float* __restrict__ wq, const float* __restrict__ wk,
                     const float* __restrict__ wv,
                     const float* __restrict__ bq, const float* __restrict__ bk,
                     const float* __restrict__ bv,
                     bf16_t* __restrict__ qh, bf16_t* __restrict__ kh,
                     bf16_t* __restrict__ vht) {
  __shared__ bf16_t ldsA[BM * LDT];
  __shared__ bf16_t ldsB[BN * LDT];
  const int z = blockIdx.z;
  const float* A    = (z == 0) ? q  : (z == 1) ? k  : v;
  const float* W    = (z == 0) ? wq : (z == 1) ? wk : wv;
  const float* bias = (z == 0) ? bq : (z == 1) ? bk : bv;
  const int m0 = blockIdx.y * BM;
  const int n0 = blockIdx.x * BN;

  v8f acc[4] = ACC_ZERO;
  for (int k0 = 0; k0 < D_MODEL; k0 += BK) {
    stage_rowmajor(ldsA, A, D_MODEL, m0, k0);
    stage_transpose(ldsB, W, D_MODEL, k0, n0);
    __syncthreads();
    wave_mma(ldsA, ldsB, acc);
    __syncthreads();
  }

  EPILOGUE_VARS
#pragma unroll
  for (int i = 0; i < 2; ++i)
#pragma unroll
    for (int j = 0; j < 2; ++j) {
      v8f c = acc[i * 2 + j];
#pragma unroll
      for (int rr = 0; rr < 8; ++rr) {
        const int m = m0 + wm + 16 * i + rr + 8 * halfc;
        const int n = n0 + wn + 16 * j + nl;
        const bf16_t val = (bf16_t)(c[rr] + bias[n]);
        const int b = m >> 11, s = m & (SEQ - 1);
        const int h = n >> 6, d = n & (DEPTH - 1);
        if (z < 2) {
          bf16_t* dst = (z == 0) ? qh : kh;
          dst[(((size_t)(b * NUM_HEADS + h)) * SEQ + s) * DEPTH + d] = val;
        } else {
          vht[(((size_t)(b * NUM_HEADS + h)) * DEPTH + d) * SEQ + s] = val;
        }
      }
    }
}

// ---------------------------------------------------------------------------
// Kernel 2 (fully fused attention):
//   logits strip (16 x SEQ) = scale*Qh_strip@Kh^T + mask*(-1e9)   (WMMA, LDS)
//   row softmax in LDS; attn strip streamed to HBM once (nontemporal)
//   ctx strip (16 x 64)   = P @ Vh                                 (WMMA)
// grid = (SEQ/16, B*H), 128 threads = 4 waves.
// ---------------------------------------------------------------------------
__global__ __launch_bounds__(NTHREADS)
void attn_fused_kernel(const bf16_t* __restrict__ qh, const bf16_t* __restrict__ kh,
                       const bf16_t* __restrict__ vht, const float* __restrict__ mask,
                       float* __restrict__ attn, float* __restrict__ ctx) {
  extern __shared__ char smem[];
  float*  logits  = (float*)smem;                                       // 16 x LSTR
  bf16_t* ldsQ    = (bf16_t*)(smem + 16 * LSTR * 4);                    // 16 x QKSTR
  bf16_t* ldsK    = (bf16_t*)(smem + 16 * LSTR * 4 + 16 * QKSTR * 2);   // 64 x QKSTR
  float*  red     = (float*)(smem + 16 * LSTR * 4 + 16 * QKSTR * 2 + 64 * QKSTR * 2);
  float*  rowstat = red + 128;                                          // 16

  const int t  = threadIdx.x;
  const int bh = blockIdx.y;
  const int b  = bh >> 4;
  const int m0 = blockIdx.x * 16;

  const bf16_t* Q = qh + ((size_t)bh * SEQ + m0) * DEPTH;
  const unsigned long long* Qg = (const unsigned long long*)Q;          // 16 ull per row
  const unsigned long long* Kg = (const unsigned long long*)(kh + (size_t)bh * SEQ * DEPTH);
  const unsigned long long* Vg = (const unsigned long long*)(vht + (size_t)bh * DEPTH * SEQ);

  // Stage Q strip (16 x 64 bf16) once.
#pragma unroll
  for (int e = 0; e < 2; ++e) {
    const int idx = t * 2 + e;        // 0..255
    const int m = idx >> 4, u = idx & 15;
    *(unsigned long long*)((char*)ldsQ + m * (QKSTR * 2) + u * 8) = Qg[m * 16 + u];
  }
  __syncthreads();

  const int lane = t & 31, wave = t >> 5;
  const int rl = lane & 15, half = lane >> 4;
  const v16bf a0 = load_frag(ldsQ + rl * QKSTR, half);       // invariant over chunks
  const v16bf a1 = load_frag(ldsQ + rl * QKSTR + 32, half);
  const int tn = wave * 16;
  const float scale = 0.125f;         // 1/sqrt(64)

  // ---- Phase 1: logits strip into LDS ----
  for (int c = 0; c < SEQ / 64; ++c) {
    const int n0 = c * 64;
    // Stage K chunk (64 rows x 64 bf16) as [n][k] via async global->LDS.
#pragma unroll
    for (int e = 0; e < 8; ++e) {
      const int idx = t * 8 + e;      // 0..1023
      const int n = idx >> 4, u = idx & 15;
      cp_b64_to_lds((char*)ldsK + n * (QKSTR * 2) + u * 8, &Kg[(size_t)(n0 + n) * 16 + u]);
    }
    cp_to_lds_wait();
    __syncthreads();

    v16bf b0 = load_frag(ldsK + (tn + rl) * QKSTR, half);
    v16bf b1 = load_frag(ldsK + (tn + rl) * QKSTR + 32, half);
    v8f acc = {0, 0, 0, 0, 0, 0, 0, 0};
    acc = __builtin_amdgcn_wmma_f32_16x16x32_bf16(false, a0, false, b0, (short)0, acc, false, false);
    acc = __builtin_amdgcn_wmma_f32_16x16x32_bf16(false, a1, false, b1, (short)0, acc, false, false);

    const int ncol = n0 + tn + rl;
    const float mk = mask[b * SEQ + ncol] * (-1e9f);
#pragma unroll
    for (int rr = 0; rr < 8; ++rr)
      logits[(rr + 8 * half) * LSTR + ncol] = acc[rr] * scale + mk;
    __syncthreads();
  }

  // ---- Phase 2: row softmax stats (exp left unnormalized in LDS) ----
  const int r = t >> 3, g = t & 7;    // 8 threads per row
  float lm = -3.0e38f;
  for (int cidx = g; cidx < SEQ; cidx += 8)
    lm = fmaxf(lm, logits[r * LSTR + cidx]);
  red[t] = lm;
  __syncthreads();
  if (t < 16) {
    float mm = red[t * 8];
#pragma unroll
    for (int i2 = 1; i2 < 8; ++i2) mm = fmaxf(mm, red[t * 8 + i2]);
    rowstat[t] = mm;
  }
  __syncthreads();
  const float mrow = rowstat[r];
  float ls = 0.0f;
  for (int cidx = g; cidx < SEQ; cidx += 8) {
    const float e2 = __expf(logits[r * LSTR + cidx] - mrow);
    logits[r * LSTR + cidx] = e2;
    ls += e2;
  }
  red[t] = ls;
  __syncthreads();
  if (t < 16) {
    float ss = 0.0f;
#pragma unroll
    for (int i2 = 0; i2 < 8; ++i2) ss += red[t * 8 + i2];
    rowstat[t] = 1.0f / ss;
  }
  __syncthreads();

  // ---- Phase 3: stream normalized attn strip to HBM (write-only -> NT) ----
  float* Crow = attn + ((size_t)bh * SEQ + m0) * SEQ;
  for (int idx = t; idx < 16 * SEQ; idx += NTHREADS) {
    const int r2 = idx >> 11, c2 = idx & (SEQ - 1);
    __builtin_nontemporal_store(logits[r2 * LSTR + c2] * rowstat[r2],
                                &Crow[(size_t)r2 * SEQ + c2]);
  }

  // ---- Phase 4: ctx strip = P(16 x SEQ, in LDS) @ Vh(SEQ x 64) ----
  bf16_t* ldsV = ldsK;                // reuse
  v8f cacc = {0, 0, 0, 0, 0, 0, 0, 0};
  for (int k0 = 0; k0 < SEQ; k0 += 64) {
    // Stage V chunk: 64 rows (d) x 64 bf16 (k) from VhT[d][s].
#pragma unroll
    for (int e = 0; e < 8; ++e) {
      const int idx = t * 8 + e;      // 0..1023
      const int dd = idx >> 4, u = idx & 15;
      cp_b64_to_lds((char*)ldsV + dd * (QKSTR * 2) + u * 8,
                    &Vg[(size_t)dd * (SEQ / 4) + (k0 >> 2) + u]);
    }
    cp_to_lds_wait();
    __syncthreads();

    v16bf af0 = frag_from_f32(logits + rl * LSTR + k0, half);
    v16bf af1 = frag_from_f32(logits + rl * LSTR + k0 + 32, half);
    v16bf bf0 = load_frag(ldsV + (tn + rl) * QKSTR, half);
    v16bf bf1 = load_frag(ldsV + (tn + rl) * QKSTR + 32, half);
    cacc = __builtin_amdgcn_wmma_f32_16x16x32_bf16(false, af0, false, bf0, (short)0, cacc, false, false);
    cacc = __builtin_amdgcn_wmma_f32_16x16x32_bf16(false, af1, false, bf1, (short)0, cacc, false, false);
    __syncthreads();
  }

  const int h = bh & (NUM_HEADS - 1);
#pragma unroll
  for (int rr = 0; rr < 8; ++rr) {
    const int m = rr + 8 * half;                 // strip row
    const int s = m0 + m;
    const int d = tn + rl;
    ctx[((size_t)(b * SEQ + s)) * D_MODEL + h * DEPTH + d] = cacc[rr] * rowstat[m];
  }
}

// ---------------------------------------------------------------------------
// Kernel 3: out = ctx (B*S x D) @ wo + bo. grid = (D/BN, B*S/BM).
// ---------------------------------------------------------------------------
__global__ __launch_bounds__(NTHREADS)
void outproj_kernel(const float* __restrict__ ctx, const float* __restrict__ wo,
                    const float* __restrict__ bo, float* __restrict__ out) {
  __shared__ bf16_t ldsA[BM * LDT];
  __shared__ bf16_t ldsB[BN * LDT];
  const int m0 = blockIdx.y * BM;
  const int n0 = blockIdx.x * BN;

  v8f acc[4] = ACC_ZERO;
  for (int k0 = 0; k0 < D_MODEL; k0 += BK) {
    stage_rowmajor(ldsA, ctx, D_MODEL, m0, k0);
    stage_transpose(ldsB, wo, D_MODEL, k0, n0);
    __syncthreads();
    wave_mma(ldsA, ldsB, acc);
    __syncthreads();
  }

  EPILOGUE_VARS
#pragma unroll
  for (int i = 0; i < 2; ++i)
#pragma unroll
    for (int j = 0; j < 2; ++j) {
      v8f c = acc[i * 2 + j];
#pragma unroll
      for (int rr = 0; rr < 8; ++rr) {
        const int m = m0 + wm + 16 * i + rr + 8 * halfc;
        const int n = n0 + wn + 16 * j + nl;
        out[(size_t)m * D_MODEL + n] = c[rr] + bo[n];
      }
    }
}

// ---------------------------------------------------------------------------
extern "C" void kernel_launch(void* const* d_in, const int* in_sizes, int n_in,
                              void* d_out, int out_size, void* d_ws, size_t ws_size,
                              hipStream_t stream) {
  (void)in_sizes; (void)n_in; (void)out_size; (void)ws_size;
  const float* v    = (const float*)d_in[0];
  const float* k    = (const float*)d_in[1];
  const float* q    = (const float*)d_in[2];
  const float* mask = (const float*)d_in[3];
  const float* wq   = (const float*)d_in[4];
  const float* bq   = (const float*)d_in[5];
  const float* wk   = (const float*)d_in[6];
  const float* bk   = (const float*)d_in[7];
  const float* wv   = (const float*)d_in[8];
  const float* bv   = (const float*)d_in[9];
  const float* wo   = (const float*)d_in[10];
  const float* bo   = (const float*)d_in[11];

  float* out  = (float*)d_out;
  float* attn = out + (size_t)BATCH * SEQ * D_MODEL;   // d_out = [out | attn]

  const size_t per = (size_t)BATCH * NUM_HEADS * SEQ * DEPTH;  // 4M elems
  bf16_t* qh  = (bf16_t*)d_ws;
  bf16_t* kh  = qh + per;
  bf16_t* vht = kh + per;
  float*  ctx = (float*)(vht + per);                    // 24 MB bf16 + 16 MB fp32

  proj_qkv_kernel<<<dim3(D_MODEL / BN, (BATCH * SEQ) / BM, 3), NTHREADS, 0, stream>>>(
      q, k, v, wq, wk, wv, bq, bk, bv, qh, kh, vht);

  attn_fused_kernel<<<dim3(SEQ / 16, BATCH * NUM_HEADS), NTHREADS, SMEM_FUSED, stream>>>(
      qh, kh, vht, mask, attn, ctx);

  outproj_kernel<<<dim3(D_MODEL / BN, (BATCH * SEQ) / BM), NTHREADS, 0, stream>>>(
      ctx, wo, bo, out);
}
